// Attention_70007966925045
// MI455X (gfx1250) — compile-verified, matrix operand order checked by reading, not run on hardware
//
#include <hip/hip_runtime.h>

// MI455X / gfx1250 multi-head attention, bf16 WMMA pipeline.
// Workspace layout (d_ws), all bf16 (unsigned short):
//   Qbf [8,12,1024,64] | Kbf | Vbf | Ybf [8192,768]   -> 4 * 12.58 MB ~= 50.3 MB

typedef __attribute__((ext_vector_type(16))) __bf16 v16bf;
typedef __attribute__((ext_vector_type(8)))  float  v8f;
typedef __attribute__((ext_vector_type(4)))  int    v4i;

#define WMMA_BF16(a, b, c) \
  __builtin_amdgcn_wmma_f32_16x16x32_bf16(false, (a), false, (b), (short)0, (c), false, false)

#if __has_builtin(__builtin_amdgcn_global_load_async_to_lds_b128) && \
    __has_builtin(__builtin_amdgcn_s_wait_asynccnt)
#define USE_ASYNC_LDS 1
#else
#define USE_ASYNC_LDS 0
#endif

typedef __attribute__((address_space(1))) v4i* gptr_v4i;
typedef __attribute__((address_space(3))) v4i* lptr_v4i;

__device__ __forceinline__ unsigned short f2bf(float f) {
  union { float f; unsigned u; } x; x.f = f;
  unsigned u = x.u;
  u += 0x7FFFu + ((u >> 16) & 1u);       // round-to-nearest-even
  return (unsigned short)(u >> 16);
}

// Load one 16x32 bf16 WMMA fragment (A layout; B uses the same pattern on Y^T rows).
// ISA layout: lane L holds row L%16; elements j=0..7 -> k = (L/16)*8 + j,
//             elements j=8..15 -> k = 16 + (L/16)*8 + (j-8): two contiguous runs of 8.
__device__ __forceinline__ v16bf load_frag(const unsigned short* tile, int stride, int lane) {
  const int row  = lane & 15;
  const int half = lane >> 4;
  const unsigned short* p = tile + row * stride + half * 8;
  union { v16bf v; uint4 q[2]; } f;
  f.q[0] = *(const uint4*)(p);
  f.q[1] = *(const uint4*)(p + 16);
  return f.v;
}

// Copy a 64x64 bf16 tile (global row-major, row stride 64) into LDS (row stride 72).
// 128 threads; 16-byte chunks. Uses GLOBAL_LOAD_ASYNC_TO_LDS_B128 (ASYNCcnt path,
// bypasses VGPRs) when the builtin is available, else a vectorized VGPR round-trip.
__device__ __forceinline__ void copy_tile64(unsigned short* __restrict__ lds,
                                            const unsigned short* __restrict__ g, int tid) {
#if USE_ASYNC_LDS
  for (int i = tid; i < 512; i += 128) {
    const int r = i >> 3, cc = (i & 7) << 3;
    __builtin_amdgcn_global_load_async_to_lds_b128(
        (gptr_v4i)(g + (size_t)r * 64 + cc),
        (lptr_v4i)(lds + r * 72 + cc), 0, 0);
  }
  __builtin_amdgcn_s_wait_asynccnt(0);
#else
  for (int i = tid; i < 512; i += 128) {
    const int r = i >> 3, cc = (i & 7) << 3;
    *(uint4*)&lds[r * 72 + cc] = *(const uint4*)&g[(size_t)r * 64 + cc];
  }
#endif
}

// ---------------------------------------------------------------------------
// Tiled GEMM: out[m][n] = sum_k A[m][k] * W[n][k] + bias[n]
// M = 8192, N = 768, K = 768. Block tile 128x64, 256 thr = 8 waves in a 4x2
// grid; each wave computes 32x32 = 4 WMMAs per k-step (A reuse 2, B reuse 2).
// MODE 0: A fp32 (x), store bf16 to [B,H,N,d]. MODE 1: A bf16 (Y), store fp32 [M,768].
// ---------------------------------------------------------------------------
template <int MODE>
__global__ __launch_bounds__(256) void gemm768(
    const void* __restrict__ Ain, const float* __restrict__ W,
    const float* __restrict__ bias,
    unsigned short* __restrict__ outBf, float* __restrict__ outF) {
  __shared__ __align__(16) unsigned short As[128 * 40];
  __shared__ __align__(16) unsigned short Bs[64 * 40];

  const int tid  = threadIdx.x;
  const int lane = tid & 31;
  const int wave = tid >> 5;
  const int wr   = wave & 3;            // 0..3 -> 32-row strip
  const int wc   = wave >> 2;           // 0..1 -> 32-col strip
  const int m0   = blockIdx.x * 128;
  const int n0   = blockIdx.y * 64;

  const float*          Af = (const float*)Ain;
  const unsigned short* Ab = (const unsigned short*)Ain;

  v8f acc[2][2];
#pragma unroll
  for (int i = 0; i < 2; ++i)
#pragma unroll
    for (int j = 0; j < 2; ++j) acc[i][j] = (v8f){};

  for (int k0 = 0; k0 < 768; k0 += 32) {
    for (int i = tid; i < 128 * 32; i += 256) {       // A: 128x32
      const int r = i >> 5, c = i & 31;
      unsigned short av;
      if (MODE == 0) av = f2bf(Af[(size_t)(m0 + r) * 768 + k0 + c]);
      else           av = Ab[(size_t)(m0 + r) * 768 + k0 + c];
      As[r * 40 + c] = av;
    }
    for (int i = tid; i < 64 * 32; i += 256) {        // B: 64x32
      const int r = i >> 5, c = i & 31;
      Bs[r * 40 + c] = f2bf(W[(size_t)(n0 + r) * 768 + k0 + c]);
    }
    if (k0 + 32 < 768) {                              // prefetch next k-step
      if (tid < 128) {
        if (MODE == 0) __builtin_prefetch(&Af[(size_t)(m0 + tid) * 768 + k0 + 32], 0, 3);
        else           __builtin_prefetch(&Ab[(size_t)(m0 + tid) * 768 + k0 + 32], 0, 3);
      } else if (tid >= 192) {
        __builtin_prefetch(&W[(size_t)(n0 + tid - 192) * 768 + k0 + 32], 0, 3);
      }
    }
    __syncthreads();
    const v16bf a0 = load_frag(As + (wr * 32) * 40,      40, lane);
    const v16bf a1 = load_frag(As + (wr * 32 + 16) * 40, 40, lane);
    const v16bf b0 = load_frag(Bs + (wc * 32) * 40,      40, lane);
    const v16bf b1 = load_frag(Bs + (wc * 32 + 16) * 40, 40, lane);
    acc[0][0] = WMMA_BF16(a0, b0, acc[0][0]);
    acc[0][1] = WMMA_BF16(a0, b1, acc[0][1]);
    acc[1][0] = WMMA_BF16(a1, b0, acc[1][0]);
    acc[1][1] = WMMA_BF16(a1, b1, acc[1][1]);
    __syncthreads();
  }

  // C/D layout: VGPR r -> row r + 8*(lane/16), col = lane%16 within 16-col tile.
  const int half = lane >> 4, nl = lane & 15;
#pragma unroll
  for (int ia = 0; ia < 2; ++ia) {
#pragma unroll
    for (int r = 0; r < 8; ++r) {
      const int m = m0 + wr * 32 + ia * 16 + r + 8 * half;
#pragma unroll
      for (int jc = 0; jc < 2; ++jc) {
        const int   n = n0 + wc * 32 + jc * 16 + nl;
        const float v = acc[ia][jc][r] + bias[n];
        if (MODE == 0) {
          const int b = m >> 10, tok = m & 1023;
          const int h = n >> 6,  dd  = n & 63;
          outBf[(((size_t)(b * 12 + h) << 10) + tok) * 64 + dd] = f2bf(v);
        } else {
          outF[(size_t)m * 768 + n] = v;
        }
      }
    }
  }
}

// ---------------------------------------------------------------------------
// Flash-style attention for one (b, h, 64-query-row tile). 128 thr = 4 waves;
// wave w owns query rows w*16..+16. 16 kv-tiles of 64. O kept in registers
// (4 x v8f per wave). head_mask applied squared (reference applies it twice).
// ---------------------------------------------------------------------------
__global__ __launch_bounds__(128) void attn64(
    const unsigned short* __restrict__ Q, const unsigned short* __restrict__ K,
    const unsigned short* __restrict__ V, const float* __restrict__ head_mask,
    unsigned short* __restrict__ Y) {
  __shared__ __align__(16) unsigned short Qs [64 * 72];
  __shared__ __align__(16) unsigned short Ks [64 * 72];
  __shared__ __align__(16) unsigned short VTs[64 * 72];
  __shared__ __align__(16) unsigned short Ps [64 * 72];
  __shared__ __align__(16) float          Ss [64 * 68];
  __shared__ float rowM[64], rowL[64], rowC[64];

  const int qt = blockIdx.x, h = blockIdx.y, b = blockIdx.z;
  const int tid = threadIdx.x, lane = tid & 31, wave = tid >> 5;
  const int half = lane >> 4, nl = lane & 15;

  const size_t headOff = ((size_t)(b * 12 + h) << 10) * 64;
  const unsigned short* Qg = Q + headOff;
  const unsigned short* Kg = K + headOff;
  const unsigned short* Vg = V + headOff;

  copy_tile64(Qs, Qg + (size_t)qt * 64 * 64, tid);
  if (tid < 64) { rowM[tid] = -3.0e38f; rowL[tid] = 0.f; }

  v8f o[4];
#pragma unroll
  for (int i = 0; i < 4; ++i) o[i] = (v8f){};
  __syncthreads();

  for (int t = 0; t < 16; ++t) {
    copy_tile64(Ks, Kg + (size_t)t * 64 * 64, tid);    // async HBM->LDS when available
    for (int i = tid; i < 64 * 64; i += 128) {         // V transposed scatter
      const int r = i >> 6, c = i & 63;
      VTs[c * 72 + r] = Vg[(size_t)(t * 64 + r) * 64 + c];
    }
    if (t + 1 < 16) {                                  // prefetch next kv tile
      const unsigned short* nxt =
          (tid < 64 ? Kg : Vg) + (size_t)((t + 1) * 64 + (tid & 63)) * 64;
      __builtin_prefetch(nxt, 0, 3);
    }
    __syncthreads();

    // S = (Q K^T) * d^-0.5 for this wave's 16 rows x 64 kv cols
    const v16bf aQ0 = load_frag(Qs + wave * 16 * 72,      72, lane);
    const v16bf aQ1 = load_frag(Qs + wave * 16 * 72 + 32, 72, lane);
#pragma unroll
    for (int jc = 0; jc < 4; ++jc) {
      const v16bf bK0 = load_frag(Ks + jc * 16 * 72,      72, lane);
      const v16bf bK1 = load_frag(Ks + jc * 16 * 72 + 32, 72, lane);
      v8f s = {};
      s = WMMA_BF16(aQ0, bK0, s);
      s = WMMA_BF16(aQ1, bK1, s);
#pragma unroll
      for (int r = 0; r < 8; ++r)
        Ss[(wave * 16 + r + 8 * half) * 68 + jc * 16 + nl] = s[r] * 0.125f;
    }
    __syncthreads();

    // Online softmax: one thread per query row.
    if (tid < 64) {
      const int r = tid;
      float mNew = rowM[r];
      for (int j = 0; j < 64; ++j) mNew = fmaxf(mNew, Ss[r * 68 + j]);
      const float corr = __expf(rowM[r] - mNew);
      float sum = 0.f;
      for (int j = 0; j < 64; ++j) {
        const float p = __expf(Ss[r * 68 + j] - mNew);
        sum += p;
        Ps[r * 72 + j] = f2bf(p);
      }
      rowM[r] = mNew;
      rowL[r] = rowL[r] * corr + sum;
      rowC[r] = corr;
    }
    __syncthreads();

    // Rescale O by per-row correction, then O += P * V.
#pragma unroll
    for (int r = 0; r < 8; ++r) {
      const float c = rowC[wave * 16 + r + 8 * half];
      o[0][r] *= c; o[1][r] *= c; o[2][r] *= c; o[3][r] *= c;
    }
    const v16bf aP0 = load_frag(Ps + wave * 16 * 72,      72, lane);
    const v16bf aP1 = load_frag(Ps + wave * 16 * 72 + 32, 72, lane);
#pragma unroll
    for (int dc = 0; dc < 4; ++dc) {
      const v16bf bV0 = load_frag(VTs + dc * 16 * 72,      72, lane);
      const v16bf bV1 = load_frag(VTs + dc * 16 * 72 + 32, 72, lane);
      o[dc] = WMMA_BF16(aP0, bV0, o[dc]);
      o[dc] = WMMA_BF16(aP1, bV1, o[dc]);
    }
    __syncthreads();
  }

  const float hm0 = head_mask[b * 12 + h];
  const float hm  = hm0 * hm0;                 // reference applies head_mask twice
#pragma unroll
  for (int r = 0; r < 8; ++r) {
    const int   mloc = wave * 16 + r + 8 * half;
    const float inv  = hm / rowL[mloc];
    unsigned short* yrow =
        Y + (size_t)((b << 10) + qt * 64 + mloc) * 768 + h * 64;
#pragma unroll
    for (int dc = 0; dc < 4; ++dc)
      yrow[dc * 16 + nl] = f2bf(o[dc][r] * inv);
  }
}

// ---------------------------------------------------------------------------
extern "C" void kernel_launch(void* const* d_in, const int* in_sizes, int n_in,
                              void* d_out, int out_size, void* d_ws, size_t ws_size,
                              hipStream_t stream) {
  const float* x     = (const float*)d_in[0];
  const float* hmask = (const float*)d_in[1];
  const float* q_w   = (const float*)d_in[2];
  const float* q_b   = (const float*)d_in[3];
  const float* k_w   = (const float*)d_in[4];
  const float* k_b   = (const float*)d_in[5];
  const float* v_w   = (const float*)d_in[6];
  const float* v_b   = (const float*)d_in[7];
  const float* p_w   = (const float*)d_in[8];
  const float* p_b   = (const float*)d_in[9];

  const size_t elems = (size_t)8 * 12 * 1024 * 64;   // == 8192 * 768
  unsigned short* qbf = (unsigned short*)d_ws;
  unsigned short* kbf = qbf + elems;
  unsigned short* vbf = kbf + elems;
  unsigned short* ybf = vbf + elems;

  const dim3 gg(64, 12);                             // 8192/128 x 768/64 tiles
  gemm768<0><<<gg, 256, 0, stream>>>(x, q_w, q_b, qbf, nullptr);
  gemm768<0><<<gg, 256, 0, stream>>>(x, k_w, k_b, kbf, nullptr);
  gemm768<0><<<gg, 256, 0, stream>>>(x, v_w, v_b, vbf, nullptr);
  attn64<<<dim3(16, 12, 8), 128, 0, stream>>>(qbf, kbf, vbf, hmask, ybf);
  gemm768<1><<<gg, 256, 0, stream>>>(ybf, p_w, p_b, nullptr, (float*)d_out);
}
